// GAE_89275190215241
// MI455X (gfx1250) — matrix operand order repaired and checked by reading, not compiled, not run on hardware
//
#include <hip/hip_runtime.h>

#define NNODES 8192
#define MAXD   40   // DEG=32 + slack; self-loop handled explicitly

typedef __attribute__((ext_vector_type(16))) __bf16 v16bf;
typedef __attribute__((ext_vector_type(8)))  __bf16 v8bf;
typedef __attribute__((ext_vector_type(8)))  float  v8f;

// ---------------------------------------------------------------------------
// Deterministic CSR-by-dst extraction from dense adjacency (one wave per dst).
// Rank via ballot+popcount -> neighbor order is ascending src, bitwise stable.
// ---------------------------------------------------------------------------
__global__ __launch_bounds__(32)
void build_csr_kernel(const int* __restrict__ adj, int* __restrict__ nbr,
                      int* __restrict__ cnt) {
  const int dst  = blockIdx.x;
  const int lane = threadIdx.x;
  int count = 0;
  for (int base = 0; base < NNODES; base += 32) {
    const int src = base + lane;
    const int a   = adj[(size_t)src * NNODES + dst];
    const unsigned mask = __builtin_amdgcn_ballot_w32(a != 0);
    if (a != 0) {
      const int rank = count + __builtin_popcount(mask & ((1u << lane) - 1u));
      if (rank < MAXD) nbr[dst * MAXD + rank] = src;
    }
    count += __builtin_popcount(mask);
  }
  if (lane == 0) cnt[dst] = count < MAXD ? count : MAXD;
}

// ---------------------------------------------------------------------------
// H = A[N,K] @ W[K,C], f32.  blockDim.x == C, ROWS rows per block.
// ---------------------------------------------------------------------------
template<int K, int C, int ROWS>
__global__ void gemm_kernel(const float* __restrict__ A,
                            const float* __restrict__ W,
                            float* __restrict__ out) {
  __shared__ float at[ROWS][K];
  const int r0 = blockIdx.x * ROWS;
  for (int i = threadIdx.x; i < ROWS * K; i += C)
    at[i / K][i % K] = A[(size_t)(r0 + i / K) * K + (i % K)];
  __syncthreads();
  const int col = threadIdx.x;
  float acc[ROWS];
#pragma unroll
  for (int r = 0; r < ROWS; ++r) acc[r] = 0.f;
  for (int k = 0; k < K; ++k) {
    const float w = W[(size_t)k * C + col];
#pragma unroll
    for (int r = 0; r < ROWS; ++r) acc[r] = fmaf(at[r][k], w, acc[r]);
  }
#pragma unroll
  for (int r = 0; r < ROWS; ++r) out[(size_t)(r0 + r) * C + col] = acc[r];
}

// ---------------------------------------------------------------------------
// s[i] = dot(H[i], a_s), t[i] = dot(H[i], a_d).  One block (C threads) per row.
// ---------------------------------------------------------------------------
template<int C>
__global__ void attn_scores_kernel(const float* __restrict__ h,
                                   const float* __restrict__ as,
                                   const float* __restrict__ ad,
                                   float* __restrict__ s, float* __restrict__ t) {
  __shared__ float rs[C];
  __shared__ float rt[C];
  const int row = blockIdx.x, tid = threadIdx.x;
  const float p = h[(size_t)row * C + tid];
  rs[tid] = p * as[tid];
  rt[tid] = p * ad[tid];
  __syncthreads();
  for (int off = C / 2; off > 0; off >>= 1) {
    if (tid < off) { rs[tid] += rs[tid + off]; rt[tid] += rt[tid + off]; }
    __syncthreads();
  }
  if (tid == 0) { s[row] = rs[0]; t[row] = rt[0]; }
}

// ---------------------------------------------------------------------------
// Per-dst GAT aggregation: leaky_relu(0.2) scores, stable softmax over
// in-neighbors + self loop, weighted sum of H[src], +bias, relu.
// One block (C threads) per dst node.
// ---------------------------------------------------------------------------
template<int C>
__global__ void gat_agg_kernel(const float* __restrict__ h,
                               const float* __restrict__ s,
                               const float* __restrict__ t,
                               const int* __restrict__ nbr,
                               const int* __restrict__ cnt,
                               const float* __restrict__ bias,
                               float* __restrict__ out) {
  const int dst = blockIdx.x, tid = threadIdx.x;
  __shared__ int   ns[MAXD + 1];
  __shared__ float ew[MAXD + 1];
  __shared__ float inv;
  const int nc = cnt[dst];            // neighbors; index nc is the self loop
  const float tdst = t[dst];
  if (tid <= nc) {
    const int src = (tid < nc) ? nbr[dst * MAXD + tid] : dst;
    ns[tid] = src;
    const float a = s[src] + tdst;
    ew[tid] = (a > 0.f) ? a : 0.2f * a;
  }
  __syncthreads();
  if (tid == 0) {
    float m = ew[0];
    for (int j = 1; j <= nc; ++j) m = fmaxf(m, ew[j]);
    float ss = 0.f;
    for (int j = 0; j <= nc; ++j) { const float e = __expf(ew[j] - m); ew[j] = e; ss += e; }
    inv = 1.f / ss;
  }
  __syncthreads();
  float acc = 0.f;
  for (int j = 0; j <= nc; ++j) acc += ew[j] * h[(size_t)ns[j] * C + tid];
  const float o = acc * inv + bias[tid];
  out[(size_t)dst * C + tid] = fmaxf(o, 0.f);
}

// ---------------------------------------------------------------------------
// f32 -> bf16 (RNE) for the reconstruction GEMM operands.
// ---------------------------------------------------------------------------
__global__ void f32_to_bf16_kernel(const float* __restrict__ in,
                                   __bf16* __restrict__ out, int n) {
  const int i = blockIdx.x * blockDim.x + threadIdx.x;
  if (i < n) out[i] = (__bf16)in[i];
}

// ---------------------------------------------------------------------------
// recon = sigmoid(r @ r.T), r in bf16 [8192,128].
// LDS-tiled: each 256-thread block computes a 128x128 output block; K staged
// in two 64-wide halves (36 KB LDS, rows padded to 72 bf16 -> conflict-free
// 16B DS reads).  Each wave owns a 16x128 strip: 8 f32 accumulators, 32x
// v_wmma_f32_16x16x32_bf16 per block.  Global operand traffic: 64 KB per
// 64 KB output (all L2 hits, r is 2 MB) -> kernel is bound by the 268 MB
// nontemporal store, which is the roofline floor.
// ---------------------------------------------------------------------------
#define RPAD 72   // padded LDS row stride in bf16 elements (64 data + 8 pad)

__global__ __launch_bounds__(256)
void recon_wmma_kernel(const __bf16* __restrict__ rb, float* __restrict__ out) {
  __shared__ __attribute__((aligned(16))) __bf16 As[128 * RPAD];
  __shared__ __attribute__((aligned(16))) __bf16 Bs[128 * RPAD];

  const int bi = blockIdx.x >> 6;      // 64x64 grid of 128x128 blocks
  const int bj = blockIdx.x & 63;
  const int i0 = bi << 7, j0 = bj << 7;
  const int tid  = threadIdx.x;
  const int w    = tid >> 5;           // wave id -> i-strip within block
  const int lane = tid & 31;
  const int m = lane & 15, hh = lane >> 4;

  v8f acc[8];
#pragma unroll
  for (int j = 0; j < 8; ++j)
#pragma unroll
    for (int v = 0; v < 8; ++v) acc[j][v] = 0.f;

#pragma unroll
  for (int ks = 0; ks < 128; ks += 64) {
    __syncthreads();                   // protect LDS reuse across stages
    // cooperative stage: 128 rows x 64 bf16 per matrix, 16B chunks
    for (int v = tid; v < 1024; v += 256) {
      const int row = v >> 3;          // 8 chunks of 8 bf16 per row
      const int ko  = (v & 7) << 3;
      *(v8bf*)(As + row * RPAD + ko) =
          *(const v8bf*)(rb + (size_t)(i0 + row) * 128 + ks + ko);
      *(v8bf*)(Bs + row * RPAD + ko) =
          *(const v8bf*)(rb + (size_t)(j0 + row) * 128 + ks + ko);
    }
    __syncthreads();

    const __bf16* arow = As + (16 * w + m) * RPAD;
#pragma unroll
    for (int kk = 0; kk < 64; kk += 32) {
      // A layout: lane (m,hh): e<8 -> K=kk+8*hh+e ; e>=8 -> K=kk+16+8*hh+e-8
      const v8bf alo = *(const v8bf*)(arow + kk + 8 * hh);
      const v8bf ahi = *(const v8bf*)(arow + kk + 16 + 8 * hh);
      const v16bf a = __builtin_shufflevector(alo, ahi,
          0, 1, 2, 3, 4, 5, 6, 7, 8, 9, 10, 11, 12, 13, 14, 15);
#pragma unroll
      for (int j = 0; j < 8; ++j) {
        // B layout: lane (m,hh): e<8 -> K=kk+8*hh+e ; e>=8 -> K=kk+16+8*hh+e-8 (two 16B DS reads)
        const v8bf blo = *(const v8bf*)(Bs + (16 * j + m) * RPAD + kk + 8 * hh);
        const v8bf bhi = *(const v8bf*)(Bs + (16 * j + m) * RPAD + kk + 16 + 8 * hh);
        const v16bf b = __builtin_shufflevector(blo, bhi,
            0, 1, 2, 3, 4, 5, 6, 7, 8, 9, 10, 11, 12, 13, 14, 15);
        acc[j] = __builtin_amdgcn_wmma_f32_16x16x32_bf16(
            false, a, false, b, (short)0, acc[j], false, false);
      }
    }
  }

  // epilogue: sigmoid via v_rcp_f32, nontemporal store (268 MB, write-once)
#pragma unroll
  for (int j = 0; j < 8; ++j) {
#pragma unroll
    for (int v = 0; v < 8; ++v) {
      const int row = i0 + 16 * w + v + 8 * hh;
      const int col = j0 + 16 * j + m;
      const float sg = __builtin_amdgcn_rcpf(1.0f + __expf(-acc[j][v]));
      __builtin_nontemporal_store(sg, out + (size_t)row * NNODES + col);
    }
  }
}

// ---------------------------------------------------------------------------
extern "C" void kernel_launch(void* const* d_in, const int* in_sizes, int n_in,
                              void* d_out, int out_size, void* d_ws, size_t ws_size,
                              hipStream_t stream) {
  (void)in_sizes; (void)n_in; (void)out_size; (void)ws_size;
  const float* x   = (const float*)d_in[0];
  const int*   adj = (const int*)d_in[1];
  // params in setup_inputs() insertion order: c1, c2, ed, d1, d2; each W,a_s,a_d,b
  const float *W1  = (const float*)d_in[2],  *as1 = (const float*)d_in[3],
              *ad1 = (const float*)d_in[4],  *b1  = (const float*)d_in[5];
  const float *W2  = (const float*)d_in[6],  *as2 = (const float*)d_in[7],
              *ad2 = (const float*)d_in[8],  *b2  = (const float*)d_in[9];
  const float *We  = (const float*)d_in[10], *ase = (const float*)d_in[11],
              *ade = (const float*)d_in[12], *be  = (const float*)d_in[13];
  const float *Wd1 = (const float*)d_in[14], *asd1= (const float*)d_in[15],
              *add1= (const float*)d_in[16], *bd1 = (const float*)d_in[17];
  const float *Wd2 = (const float*)d_in[18], *asd2= (const float*)d_in[19],
              *add2= (const float*)d_in[20], *bd2 = (const float*)d_in[21];

  float* out   = (float*)d_out;
  float* xoout = out + (size_t)NNODES * NNODES;                  // [8192,256]
  float* zout  = xoout + (size_t)NNODES * 256;                   // [8192,256]

  char* wsb = (char*)d_ws;
  float*  Hbuf = (float*) (wsb + 0);          // [8192,256] pre-aggregation H
  float*  h1   = (float*) (wsb + 8388608);    // [8192,128] (reused as xo1)
  float*  rbuf = (float*) (wsb + 12582912);   // [8192,128]
  __bf16* rb   = (__bf16*)(wsb + 16777216);   // [8192,128] bf16
  float*  sarr = (float*) (wsb + 18874368);   // [8192]
  float*  tarr = (float*) (wsb + 18907136);   // [8192]
  int*    nbr  = (int*)   (wsb + 18939904);   // [8192,MAXD]
  int*    cnt  = (int*)   (wsb + 20250624);   // [8192]

  // ---- graph extraction (deterministic) ----
  build_csr_kernel<<<NNODES, 32, 0, stream>>>(adj, nbr, cnt);

  // ---- layer c1: x[.,256] -> h1[.,128] ----
  gemm_kernel<256, 128, 8><<<NNODES / 8, 128, 0, stream>>>(x, W1, Hbuf);
  attn_scores_kernel<128><<<NNODES, 128, 0, stream>>>(Hbuf, as1, ad1, sarr, tarr);
  gat_agg_kernel<128><<<NNODES, 128, 0, stream>>>(Hbuf, sarr, tarr, nbr, cnt, b1, h1);

  // ---- layer c2: h1[.,128] -> z[.,256] (written straight into d_out) ----
  gemm_kernel<128, 256, 8><<<NNODES / 8, 256, 0, stream>>>(h1, W2, Hbuf);
  attn_scores_kernel<256><<<NNODES, 256, 0, stream>>>(Hbuf, as2, ad2, sarr, tarr);
  gat_agg_kernel<256><<<NNODES, 256, 0, stream>>>(Hbuf, sarr, tarr, nbr, cnt, b2, zout);

  // ---- layer ed: z[.,256] -> r[.,128] ----
  gemm_kernel<256, 128, 8><<<NNODES / 8, 128, 0, stream>>>(zout, We, Hbuf);
  attn_scores_kernel<128><<<NNODES, 128, 0, stream>>>(Hbuf, ase, ade, sarr, tarr);
  gat_agg_kernel<128><<<NNODES, 128, 0, stream>>>(Hbuf, sarr, tarr, nbr, cnt, be, rbuf);

  // ---- recon = sigmoid(r @ r.T) via bf16 WMMA, LDS-tiled ----
  f32_to_bf16_kernel<<<(NNODES * 128) / 256, 256, 0, stream>>>(rbuf, rb, NNODES * 128);
  recon_wmma_kernel<<<64 * 64, 256, 0, stream>>>(rb, out);

  // ---- layer d1: z[.,256] -> xo1[.,128] (reuse h1 buffer) ----
  gemm_kernel<256, 128, 8><<<NNODES / 8, 128, 0, stream>>>(zout, Wd1, Hbuf);
  attn_scores_kernel<128><<<NNODES, 128, 0, stream>>>(Hbuf, asd1, add1, sarr, tarr);
  gat_agg_kernel<128><<<NNODES, 128, 0, stream>>>(Hbuf, sarr, tarr, nbr, cnt, bd1, h1);

  // ---- layer d2: xo1[.,128] -> xo[.,256] (into d_out) ----
  gemm_kernel<128, 256, 8><<<NNODES / 8, 256, 0, stream>>>(h1, Wd2, Hbuf);
  attn_scores_kernel<256><<<NNODES, 256, 0, stream>>>(Hbuf, asd2, add2, sarr, tarr);
  gat_agg_kernel<256><<<NNODES, 256, 0, stream>>>(Hbuf, sarr, tarr, nbr, cnt, bd2, xoout);
}
